// SpatialMemory_83872121357101
// MI455X (gfx1250) — matrix operand order repaired and checked by reading, not compiled, not run on hardware
//
#include <hip/hip_runtime.h>

typedef __attribute__((ext_vector_type(16))) _Float16 v16h;
typedef __attribute__((ext_vector_type(8)))  _Float16 v8h;
typedef __attribute__((ext_vector_type(8)))  float    v8f;
typedef __attribute__((ext_vector_type(4)))  float    v4f;

#define HIDDEN 1024
#define NMEM   50
#define SEQLEN 200

// workspace layout (bytes)
static constexpr size_t OFF_WF   = 0;                          // wf_f16 [1024][1024]
static constexpr size_t OFF_MEMP = OFF_WF   + 1024u*1024u*2u;  // mem_f16 padded [64][1024]
static constexpr size_t OFF_MEMT = OFF_MEMP + 64u*1024u*2u;    // memT_f16 [1024][64]
static constexpr size_t OFF_M2T  = OFF_MEMT + 1024u*64u*2u;    // m2t_f16 [1024][64]

__device__ __forceinline__ v16h pack16(v8h lo, v8h hi) {
  union { v16h v; v8h h[2]; } u;
  u.h[0] = lo; u.h[1] = hi;
  return u.v;
}

// ---- Precompute: f16 copies of weights/memory in WMMA-friendly layouts ----
__global__ void prep_convert(const float* __restrict__ gw,
                             const float* __restrict__ mem,
                             _Float16* __restrict__ wf,
                             _Float16* __restrict__ memp,
                             _Float16* __restrict__ memt) {
  int idx0 = blockIdx.x * blockDim.x + threadIdx.x;
  int stride = gridDim.x * blockDim.x;
  for (int i = idx0; i < 1024 * 1024; i += stride) {       // wf[o][h] = gw[o][h]
    int o = i >> 10, h = i & 1023;
    wf[i] = (_Float16)gw[o * 2048 + h];
  }
  for (int i = idx0; i < 64 * 1024; i += stride) {         // memp[m][h], rows 50..63 = 0
    int m = i >> 10, h = i & 1023;
    memp[i] = (m < NMEM) ? (_Float16)mem[m * 1024 + h] : (_Float16)0.0f;
  }
  for (int i = idx0; i < 1024 * 64; i += stride) {         // memt[h][m] = mem[m][h]
    int h = i >> 6, m = i & 63;
    memt[i] = (m < NMEM) ? (_Float16)mem[m * 1024 + h] : (_Float16)0.0f;
  }
}

// M2T[o][m] = sum_j mem[m][j] * gw[o][1024+j]   (mem @ w_m^T, transposed, f16)
__global__ void prep_m2t(const float* __restrict__ gw,
                         const float* __restrict__ mem,
                         _Float16* __restrict__ m2t) {
  int i = blockIdx.x * blockDim.x + threadIdx.x;  // 0 .. 65535
  int o = i >> 6, m = i & 63;
  float acc = 0.0f;
  if (m < NMEM) {
    const float* a = mem + m * 1024;
    const float* b = gw + o * 2048 + 1024;
    for (int j = 0; j < 1024; j += 4) {
      v4f av = *(const v4f*)(a + j);
      v4f bv = *(const v4f*)(b + j);
      acc += av.x * bv.x + av.y * bv.y + av.z * bv.z + av.w * bv.w;
    }
  }
  m2t[i] = (_Float16)acc;
}

// ---- Fused main kernel: 32 rows per block (2 M-tiles), 8 waves ----
__global__ __launch_bounds__(256, 1) void spatial_memory_fused(
    const float* __restrict__ x, const float* __restrict__ pos,
    const float* __restrict__ gate_b,
    const _Float16* __restrict__ wf, const _Float16* __restrict__ memp,
    const _Float16* __restrict__ memt, const _Float16* __restrict__ m2t,
    float* __restrict__ out) {
  __shared__ __attribute__((aligned(16))) _Float16 sFeats[32 * 1032]; // 66,048 B
  __shared__ __attribute__((aligned(16))) _Float16 sW[32 * 72];       //  4,608 B
  __shared__ __attribute__((aligned(16))) float    sSim[32 * 68];     //  8,704 B

  const int tid  = threadIdx.x;
  const int lane = tid & 31;
  const int wave = tid >> 5;
  const int row0 = blockIdx.x * 32;

  const int arow  = lane & 15;        // A row / B column within a 16-tile
  const int ahalf = lane >> 4;        // 0 or 1
  const int c0    = ahalf * 8;        // A-fragment chunk base (elements)
  const int kbB   = ahalf * 8;        // B-fragment k base (elements)

  // Phase 1: feats = x + pos  ->  f16 LDS (32 x 1024)
  for (int i = tid; i < 32 * 256; i += 256) {
    int r = i >> 8;
    int col = (i & 255) * 4;
    int n = row0 + r;
    int s = n % SEQLEN;
    v4f xv = *(const v4f*)(x + (size_t)n * HIDDEN + col);
    v4f pv = *(const v4f*)(pos + (size_t)s * HIDDEN + col);
    _Float16* dst = &sFeats[r * 1032 + col];
    dst[0] = (_Float16)(xv.x + pv.x); dst[1] = (_Float16)(xv.y + pv.y);
    dst[2] = (_Float16)(xv.z + pv.z); dst[3] = (_Float16)(xv.w + pv.w);
  }
  __syncthreads();

  // Phase 2: sim = feats @ mem^T  (32 x 64, cols >= 50 zero) — all 8 waves,
  // two interleaved accumulators to break the dependent-WMMA chain
  {
    const int mt = wave >> 2;         // M-tile 0/1
    const int nt = wave & 3;          // N-tile 0..3 over the 64 memory slots
    v8f acc0 = {}, acc1 = {};
    const _Float16* ap = &sFeats[(mt * 16 + arow) * 1032];
    const _Float16* bp = memp + (size_t)(nt * 16 + arow) * HIDDEN;
    for (int k = 0; k < 32; k += 2) {
      const _Float16* a0 = ap + k * 32 + c0;
      const _Float16* b0 = bp + k * 32 + kbB;
      v16h A0 = pack16(*(const v8h*)a0, *(const v8h*)(a0 + 16));
      v16h B0 = pack16(*(const v8h*)b0, *(const v8h*)(b0 + 16));
      acc0 = __builtin_amdgcn_wmma_f32_16x16x32_f16(false, A0, false, B0,
                                                    (short)0, acc0, false, false);
      const _Float16* a1 = a0 + 32;
      const _Float16* b1 = b0 + 32;
      v16h A1 = pack16(*(const v8h*)a1, *(const v8h*)(a1 + 16));
      v16h B1 = pack16(*(const v8h*)b1, *(const v8h*)(b1 + 16));
      acc1 = __builtin_amdgcn_wmma_f32_16x16x32_f16(false, A1, false, B1,
                                                    (short)0, acc1, false, false);
    }
#pragma unroll
    for (int v = 0; v < 8; ++v)
      sSim[(mt * 16 + v + ahalf * 8) * 68 + nt * 16 + arow] = acc0[v] + acc1[v];
  }
  __syncthreads();

  // Phase 3: softmax over the 50 valid columns -> w (f16, zero-padded to 64)
  if (tid < 32) {
    float mx = -1e30f;
    for (int m = 0; m < NMEM; ++m) mx = fmaxf(mx, sSim[tid * 68 + m]);
    float sum = 0.0f;
    for (int m = 0; m < NMEM; ++m) sum += __expf(sSim[tid * 68 + m] - mx);
    float inv = 1.0f / sum;
    for (int m = 0; m < NMEM; ++m)
      sW[tid * 72 + m] = (_Float16)(__expf(sSim[tid * 68 + m] - mx) * inv);
    for (int m = NMEM; m < 64; ++m) sW[tid * 72 + m] = (_Float16)0.0f;
  }
  __syncthreads();

  // Phase 4: two N-passes of 4 N-tiles x 2 M-tiles per wave.
  // 128 accumulator VGPRs -> stays inside the 256-VGPR window (no MSB thrash);
  // B fragments still reused across both M-tiles.
  const v8f zf = {0.f, 0.f, 0.f, 0.f, 0.f, 0.f, 0.f, 0.f};
  for (int np = 0; np < 2; ++np) {
    const int colBase = wave * 128 + np * 64;
    v8f Cg[2][4], Cmr[2][4];
#pragma unroll
    for (int i = 0; i < 4; ++i) {
      Cg[0][i] = zf; Cg[1][i] = zf; Cmr[0][i] = zf; Cmr[1][i] = zf;
    }

    // mem_read = w @ mem   (K = 64, B = memt[h][m])
#pragma unroll
    for (int k2 = 0; k2 < 2; ++k2) {
      const _Float16* a0 = &sW[arow * 72 + k2 * 32 + c0];
      const _Float16* a1 = &sW[(16 + arow) * 72 + k2 * 32 + c0];
      v16h A0 = pack16(*(const v8h*)a0, *(const v8h*)(a0 + 16));
      v16h A1 = pack16(*(const v8h*)a1, *(const v8h*)(a1 + 16));
#pragma unroll
      for (int i = 0; i < 4; ++i) {
        const _Float16* b0 = memt + (size_t)(colBase + i * 16 + arow) * 64 + k2 * 32 + kbB;
        v16h B = pack16(*(const v8h*)b0, *(const v8h*)(b0 + 16));
        Cmr[0][i] = __builtin_amdgcn_wmma_f32_16x16x32_f16(false, A0, false, B,
                                                           (short)0, Cmr[0][i], false, false);
        Cmr[1][i] = __builtin_amdgcn_wmma_f32_16x16x32_f16(false, A1, false, B,
                                                           (short)0, Cmr[1][i], false, false);
      }
    }

    // gate logits: feats @ w_f^T   (K = 1024, B = wf[o][h])
    for (int k = 0; k < 32; ++k) {
      const _Float16* a0 = &sFeats[arow * 1032 + k * 32 + c0];
      const _Float16* a1 = &sFeats[(16 + arow) * 1032 + k * 32 + c0];
      v16h A0 = pack16(*(const v8h*)a0, *(const v8h*)(a0 + 16));
      v16h A1 = pack16(*(const v8h*)a1, *(const v8h*)(a1 + 16));
#pragma unroll
      for (int i = 0; i < 4; ++i) {
        const _Float16* b0 = wf + (size_t)(colBase + i * 16 + arow) * HIDDEN + k * 32 + kbB;
        v16h B = pack16(*(const v8h*)b0, *(const v8h*)(b0 + 16));
        Cg[0][i] = __builtin_amdgcn_wmma_f32_16x16x32_f16(false, A0, false, B,
                                                          (short)0, Cg[0][i], false, false);
        Cg[1][i] = __builtin_amdgcn_wmma_f32_16x16x32_f16(false, A1, false, B,
                                                          (short)0, Cg[1][i], false, false);
      }
    }

    // + w @ M2  (mem_read @ w_m^T folded through rank-50 factor; K = 64)
#pragma unroll
    for (int k2 = 0; k2 < 2; ++k2) {
      const _Float16* a0 = &sW[arow * 72 + k2 * 32 + c0];
      const _Float16* a1 = &sW[(16 + arow) * 72 + k2 * 32 + c0];
      v16h A0 = pack16(*(const v8h*)a0, *(const v8h*)(a0 + 16));
      v16h A1 = pack16(*(const v8h*)a1, *(const v8h*)(a1 + 16));
#pragma unroll
      for (int i = 0; i < 4; ++i) {
        const _Float16* b0 = m2t + (size_t)(colBase + i * 16 + arow) * 64 + k2 * 32 + kbB;
        v16h B = pack16(*(const v8h*)b0, *(const v8h*)(b0 + 16));
        Cg[0][i] = __builtin_amdgcn_wmma_f32_16x16x32_f16(false, A0, false, B,
                                                          (short)0, Cg[0][i], false, false);
        Cg[1][i] = __builtin_amdgcn_wmma_f32_16x16x32_f16(false, A1, false, B,
                                                          (short)0, Cg[1][i], false, false);
      }
    }

    // Epilogue for this pass: out = feats + sigmoid(Cg + b) * mem_read
#pragma unroll
    for (int i = 0; i < 4; ++i) {
      int col = colBase + i * 16 + arow;
      float gb = gate_b[col];
#pragma unroll
      for (int mt = 0; mt < 2; ++mt) {
#pragma unroll
        for (int v = 0; v < 8; ++v) {
          int n = row0 + mt * 16 + v + ahalf * 8;
          int s = n % SEQLEN;
          float g = 1.0f / (1.0f + __expf(-(Cg[mt][i][v] + gb)));
          float f = x[(size_t)n * HIDDEN + col] + pos[(size_t)s * HIDDEN + col];
          out[(size_t)n * HIDDEN + col] = f + g * Cmr[mt][i][v];
        }
      }
    }
  }
}

extern "C" void kernel_launch(void* const* d_in, const int* in_sizes, int n_in,
                              void* d_out, int out_size, void* d_ws, size_t ws_size,
                              hipStream_t stream) {
  const float* x   = (const float*)d_in[0];
  const float* mem = (const float*)d_in[1];
  const float* pos = (const float*)d_in[2];
  const float* gw  = (const float*)d_in[3];
  const float* gb  = (const float*)d_in[4];
  char* ws = (char*)d_ws;
  _Float16* wf   = (_Float16*)(ws + OFF_WF);
  _Float16* memp = (_Float16*)(ws + OFF_MEMP);
  _Float16* memt = (_Float16*)(ws + OFF_MEMT);
  _Float16* m2t  = (_Float16*)(ws + OFF_M2T);

  prep_convert<<<512, 256, 0, stream>>>(gw, mem, wf, memp, memt);
  prep_m2t<<<256, 256, 0, stream>>>(gw, mem, m2t);
  spatial_memory_fused<<<1600, 256, 0, stream>>>(x, pos, gb, wf, memp, memt, m2t,
                                                 (float*)d_out);
}
